// CDCL_47906065219864
// MI455X (gfx1250) — matrix operand order repaired and checked by reading, not compiled, not run on hardware
//
#include <hip/hip_runtime.h>
#include <hip/hip_bf16.h>

// ---------------------------------------------------------------------------
// CDNA5 (gfx1250) implementation:
//  - extractor convs as implicit GEMM: bf16 im2col + v_wmma_f32_16x16x32_bf16
//    with fragment-major pre-swizzled B (L2-resident). LDS-free, software-
//    pipelined inner loop: fragments for K-step i+1 load while the WMMA for
//    step i issues, so s_wait covers older loads only.
//  - contrastive loss as a fused flash-style WMMA GEMM with online softmax
//    reductions (never materializes the 6400x12800 logits); the shared feat
//    tile is double-buffered in LDS via CDNA5 async copies
//    (global_load_async_to_lds_b128 / s_wait_asynccnt) so the DMA of tile
//    n+1 overlaps the WMMA compute of tile n.
//  - wave32-aware shuffles / layouts throughout.
// ---------------------------------------------------------------------------

typedef __attribute__((ext_vector_type(16))) __bf16 v16bf;
typedef __attribute__((ext_vector_type(8)))  float  v8f;

union FragBF { uint4 u[2]; v16bf v; };   // 16 bf16 halves = two 16B chunks

__device__ __forceinline__ int imin(int a, int b) { return a < b ? a : b; }
__device__ __forceinline__ int imax(int a, int b) { return a > b ? a : b; }

__device__ __forceinline__ float wave_sum(float v) {
    #pragma unroll
    for (int m = 16; m >= 1; m >>= 1) v += __shfl_xor(v, m, 32);
    return v;
}

// map pixel-major index m = b*1600 + y*40 + x  ->  patch-order feature row
__device__ __forceinline__ int nmap(int m) {
    int b = m / 1600, rem = m % 1600;
    int y = rem / 40, x = rem % 40;
    int py = y / 10, iy = y % 10, px = x / 10, ix = x % 10;
    return ((py * 4 + px) * 4 + b) * 100 + iy * 10 + ix;
}

// fragment-major index for the packed 12800x128 feature bank:
// [nb = n>>4][kt = k>>5][n&15][k&31]
__device__ __forceinline__ size_t featp_idx(int n, int k) {
    return ((size_t)(n >> 4) * 4 + (k >> 5)) * 512 + (size_t)(n & 15) * 32 + (k & 31);
}

// ---------------------------------------------------------------------------
// zero accumulators
// ---------------------------------------------------------------------------
__global__ void k_zero(float* accum, int* cntAll, float* patchSum, float* patchLab) {
    int t = threadIdx.x;
    if (t < 8)  accum[t] = 0.f;
    if (t < 2)  cntAll[t] = 0;
    if (t < 64) { patchSum[t] = 0.f; patchLab[t] = 0.f; }
}

// ---------------------------------------------------------------------------
// pack conv/proj weight OIHW (N x Kreal) -> fragment-major bf16:
// [kb = k>>5][nb = n>>4][n&15][k&31], K padded with zeros
// ---------------------------------------------------------------------------
__global__ void k_pack_wt(const float* __restrict__ W, __bf16* __restrict__ Wt,
                          int N, int K, int Kreal) {
    int idx = blockIdx.x * 256 + threadIdx.x;
    if (idx >= N * K) return;
    int n = idx % N, k = idx / N;
    float v = (k < Kreal) ? W[(size_t)n * Kreal + k] : 0.f;
    size_t dst = ((size_t)(k >> 5) * (N >> 4) + (n >> 4)) * 512
               + (size_t)(n & 15) * 32 + (k & 31);
    Wt[dst] = (__bf16)v;
}

// ---------------------------------------------------------------------------
// im2col (stride 2, 3x3, SAME with pad_low=0) -> bf16 A[M][K]
// nchw=1: input NCHW; nchw=0: input pixel-major (b,y,x)*C + c
// ---------------------------------------------------------------------------
__global__ void k_im2col(const float* __restrict__ in, __bf16* __restrict__ out,
                         int B, int Cin, int Hin, int Win, int Hout, int Wout,
                         int K, int Kreal, int nchw) {
    size_t idx = blockIdx.x * (size_t)256 + threadIdx.x;
    size_t total = (size_t)B * Hout * Wout * K;
    if (idx >= total) return;
    int k = (int)(idx % K);
    size_t m = idx / K;
    int x = (int)(m % Wout);
    int y = (int)((m / Wout) % Hout);
    int b = (int)(m / ((size_t)Wout * Hout));
    float v = 0.f;
    if (k < Kreal) {
        int i = k / 9, rem = k % 9, dy = rem / 3, dx = rem % 3;
        int yy = 2 * y + dy, xx = 2 * x + dx;
        if (yy < Hin && xx < Win) {
            v = nchw ? in[(((size_t)b * Cin + i) * Hin + yy) * Win + xx]
                     : in[(((size_t)b * Hin + yy) * Win + xx) * Cin + i];
        }
    }
    out[idx] = (__bf16)v;
}

// ---------------------------------------------------------------------------
// WMMA GEMM: C[M][N] (f32, pixel-major) = (relu)(A[M][K]bf16 @ Bp)
// Bp is fragment-major packed (see k_pack_wt) and L2-resident: no LDS, no
// barriers. Software-pipelined: loads for K-step i+1 issue before the WMMA
// of step i. N/K are compile-time so stores/strides use immediate offsets.
// 1 wave per 16x16 tile, block = 4 waves over M.
// (The tail loads read one 32-half tile past A/Bp; carves are padded.)
// ---------------------------------------------------------------------------
template <int N, int K, bool RELU>
__global__ void k_gemm_wmma(const __bf16* __restrict__ A, const __bf16* __restrict__ Bp,
                            float* __restrict__ C) {
    int lane  = threadIdx.x & 31;
    int wave  = threadIdx.x >> 5;
    int mbase = (blockIdx.x * 4 + wave) * 16;
    int nbIdx = blockIdx.y;
    int mrow  = mbase + (lane & 15);
    int g     = lane >> 4;            // wave32 half select

    const __bf16* arow = A + (size_t)mrow * K;
    const __bf16* bptr = Bp + (size_t)nbIdx * 512 + (size_t)(lane & 15) * 32;
    constexpr size_t bstride = (size_t)(N / 16) * 512;

    FragBF fa, fb;
    fa.u[0] = *(const uint4*)(arow + g * 8);        // halves 0..7  (K g*8..)
    fa.u[1] = *(const uint4*)(arow + 16 + g * 8);   // halves 8..15 (K 16+g*8..)
    fb.u[0] = *(const uint4*)(bptr + g * 8);        // halves 0..7  (K g*8..)
    fb.u[1] = *(const uint4*)(bptr + 16 + g * 8);   // halves 8..15

    v8f acc = {};
    #pragma unroll 4
    for (int k0 = 32; k0 <= K; k0 += 32) {
        FragBF na, nb2;
        const __bf16* an = arow + k0;
        const __bf16* bn = bptr + (size_t)(k0 >> 5) * bstride;
        na.u[0]  = *(const uint4*)(an + g * 8);
        na.u[1]  = *(const uint4*)(an + 16 + g * 8);
        nb2.u[0] = *(const uint4*)(bn + g * 8);
        nb2.u[1] = *(const uint4*)(bn + 16 + g * 8);
        acc = __builtin_amdgcn_wmma_f32_16x16x32_bf16(false, fa.v, false, fb.v,
                                                      (short)0, acc, false, false);
        fa = na; fb = nb2;
    }

    int nn   = nbIdx * 16 + (lane & 15);
    int mofs = g * 8;                  // lanes 16-31 hold rows r+8
    float* cp = C + (size_t)(mbase + mofs) * N + nn;
    #pragma unroll
    for (int r = 0; r < 8; ++r) {
        float v = acc[r];
        if (RELU) v = fmaxf(v, 0.f);
        cp[(size_t)r * N] = v;
    }
}

// ---------------------------------------------------------------------------
// N=2 classifier head: Z[m][2] = FEA[m][:]@Wc^T + bc
// ---------------------------------------------------------------------------
__global__ void k_head2(const float* __restrict__ F, const float* __restrict__ Wc,
                        const float* __restrict__ bc, float* __restrict__ Z, int M) {
    int m = blockIdx.x * 256 + threadIdx.x;
    if (m >= M) return;
    float a0 = bc[0], a1 = bc[1];
    const float* f = F + (size_t)m * 256;
    for (int k = 0; k < 256; ++k) {
        float x = f[k];
        a0 += x * Wc[k];
        a1 += x * Wc[256 + k];
    }
    Z[(size_t)m * 2 + 0] = a0;
    Z[(size_t)m * 2 + 1] = a1;
}

// ---------------------------------------------------------------------------
// f32 -> bf16 copy (pixel-major FEA for the projection GEMM)
// ---------------------------------------------------------------------------
__global__ void k_f2bf(const float* __restrict__ in, __bf16* __restrict__ out, size_t n) {
    size_t i = blockIdx.x * (size_t)256 + threadIdx.x;
    if (i < n) out[i] = (__bf16)in[i];
}

// ---------------------------------------------------------------------------
// L2-normalize proj rows; write patch-order bf16 anchors (row-major, for the
// A operand) + packed fragment-major feature bank (s-half)
// ---------------------------------------------------------------------------
__global__ void k_norm_pack(const float* __restrict__ P, __bf16* __restrict__ Anc,
                            __bf16* __restrict__ FeatP) {
    int m = blockIdx.x * 256 + threadIdx.x;
    if (m >= 6400) return;
    const float* row = P + (size_t)m * 128;
    float s = 0.f;
    for (int k = 0; k < 128; ++k) s += row[k] * row[k];
    float inv = 1.f / fmaxf(sqrtf(s), 1e-12f);
    int n = nmap(m);
    for (int k = 0; k < 128; ++k) {
        float v = row[k] * inv;
        Anc[(size_t)n * 128 + k] = (__bf16)v;
        FeatP[featp_idx(n, k)]   = (__bf16)v;
    }
}

// EMA projections (input, NCHW) -> packed feature bank (e-half, patch order)
__global__ void k_pack_ema(const float* __restrict__ E, __bf16* __restrict__ FeatP) {
    int m = blockIdx.x * 256 + threadIdx.x;
    if (m >= 6400) return;
    int b = m / 1600, y = (m / 40) % 40, x = m % 40;
    int n = 6400 + nmap(m);
    for (int k = 0; k < 128; ++k)
        FeatP[featp_idx(n, k)] =
            (__bf16)E[(((size_t)b * 128 + k) * 40 + y) * 40 + x];
}

// ---------------------------------------------------------------------------
// unlabeled stats: entropy, pseudo-labels, EMA confidence, consistency CE,
// class histogram, per-patch label fraction
// ---------------------------------------------------------------------------
__global__ void k_z_stats(const float* __restrict__ Z, const float* __restrict__ Zema,
                          int* __restrict__ labAll, int* __restrict__ cntAll,
                          float* __restrict__ patchLab, float* __restrict__ accum) {
    int m = blockIdx.x * 256 + threadIdx.x;   // exact grid: 6400
    float z0 = Z[(size_t)m * 2 + 0], z1 = Z[(size_t)m * 2 + 1];
    float mx = fmaxf(z0, z1);
    float lse = mx + __logf(__expf(z0 - mx) + __expf(z1 - mx));
    float lp0 = z0 - lse, lp1 = z1 - lse;
    float p0 = __expf(lp0), p1 = __expf(lp1);
    float ent = -(p0 * lp0 + p1 * lp1);
    int pl = (z1 > z0) ? 1 : 0;

    int b = m / 1600, y = (m / 40) % 40, x = m % 40;
    float e0 = Zema[(((size_t)b * 2 + 0) * 40 + y) * 40 + x];
    float e1 = Zema[(((size_t)b * 2 + 1) * 40 + y) * 40 + x];
    float me = fmaxf(e0, e1);
    float se = __expf(e0 - me) + __expf(e1 - me);
    float pmax = 1.f / se;                   // exp(max-max)/sum
    int ple = (e1 > e0) ? 1 : 0;
    bool conf = pmax > 0.6f;
    float cnll = conf ? (ple ? -lp1 : -lp0) : 0.f;
    float ccnt = conf ? 1.f : 0.f;

    float r0 = wave_sum(ent), r1 = wave_sum(cnll), r2 = wave_sum(ccnt);
    if ((threadIdx.x & 31) == 0) {
        atomicAdd(&accum[0], r0);
        atomicAdd(&accum[1], r1);
        atomicAdd(&accum[2], r2);
    }
    int n = nmap(m);
    labAll[n]        = pl;
    labAll[6400 + n] = ple;
    atomicAdd(&cntAll[pl], 1);
    atomicAdd(&cntAll[ple], 1);
    atomicAdd(&patchLab[n / 100], (float)pl);
}

// ---------------------------------------------------------------------------
// supervised CE with fused bilinear upsample 40->320 of zl_small
// ---------------------------------------------------------------------------
__global__ void k_sup_loss(const float* __restrict__ Zs, const int* __restrict__ yl,
                           float* __restrict__ accum) {
    int idx = blockIdx.x * 256 + threadIdx.x;  // exact grid: 409600
    int b = idx / 102400;
    int rem = idx % 102400;
    int y = rem / 320, x = rem % 320;
    const float sc = 39.0f / 319.0f;
    float fy = y * sc, fx = x * sc;
    int y0 = (int)fy, x0 = (int)fx;
    int y1 = imin(y0 + 1, 39), x1 = imin(x0 + 1, 39);
    float wy = fy - (float)y0, wx = fx - (float)x0;
    int base = b * 1600;
    float z[2];
    #pragma unroll
    for (int c = 0; c < 2; ++c) {
        float v00 = Zs[(size_t)(base + y0 * 40 + x0) * 2 + c];
        float v01 = Zs[(size_t)(base + y0 * 40 + x1) * 2 + c];
        float v10 = Zs[(size_t)(base + y1 * 40 + x0) * 2 + c];
        float v11 = Zs[(size_t)(base + y1 * 40 + x1) * 2 + c];
        float top = v00 * (1.f - wx) + v01 * wx;
        float bot = v10 * (1.f - wx) + v11 * wx;
        z[c] = top * (1.f - wy) + bot * wy;
    }
    int lab = yl[idx];
    bool valid = (lab != 255);
    int sl = imin(imax(lab, 0), 1);
    float mx = fmaxf(z[0], z[1]);
    float lse = mx + __logf(__expf(z[0] - mx) + __expf(z[1] - mx));
    float nll = -((sl ? z[1] : z[0]) - lse);
    float c1 = valid ? nll : 0.f, c2 = valid ? 1.f : 0.f;
    c1 = wave_sum(c1); c2 = wave_sum(c2);
    if ((threadIdx.x & 31) == 0) {
        atomicAdd(&accum[3], c1);
        atomicAdd(&accum[4], c2);
    }
}

// ---------------------------------------------------------------------------
// Fused contrastive loss: 6400x12800x128 WMMA GEMM with online
// sum-exp / positive-sum row reductions (flash-softmax style).
// 1 wave = 16 anchor rows; A resident in VGPRs across all 800 N-tiles.
// The fragment-major 128x16 feat tile (4 KB) is shared by the block's 4
// waves and DOUBLE-BUFFERED: the async global->LDS copy (ASYNCcnt) of tile
// n+1 is issued before computing tile n, overlapping DMA with WMMA.
// ---------------------------------------------------------------------------
__global__ void k_contr_wmma(const __bf16* __restrict__ Anc,
                             const __bf16* __restrict__ FeatP,
                             const int* __restrict__ labAll,
                             const int* __restrict__ cntAll,
                             float* __restrict__ patchSum) {
    __shared__ __bf16 ldsT[2][2048];   // 2 x (4 k-tiles x [n16][k32]), LDS offset 0
    int tid   = threadIdx.x;
    int lane  = tid & 31;
    int wave  = tid >> 5;
    int mbase = (blockIdx.x * 4 + wave) * 16;
    int mrow  = mbase + (lane & 15);
    int g     = lane >> 4;
    int mofs  = g * 8;

    // preload A fragments for all 4 k-steps (K=128); reused for 800 N-tiles
    v16bf av[4];
    #pragma unroll
    for (int t = 0; t < 4; ++t) {
        const __bf16* arow = Anc + (size_t)mrow * 128 + t * 32;
        FragBF fa;
        fa.u[0] = *(const uint4*)(arow + g * 8);
        fa.u[1] = *(const uint4*)(arow + 16 + g * 8);
        av[t] = fa.v;
    }
    int rowlab[8];
    #pragma unroll
    for (int r = 0; r < 8; ++r) rowlab[r] = labAll[mbase + mofs + r];

    int fragoff = (lane & 15) * 32;

    // async tile copy: 128 threads x 32B, no VGPR round-trip
    // (GLOBAL_LOAD_ASYNC_TO_LDS_B128, tracked by ASYNCcnt)
    auto issueTile = [&](int nb, int buf) {
        unsigned ldsOff = (unsigned)(buf * 4096 + tid * 32);   // bytes, base 0
        const __bf16* src = FeatP + (size_t)nb * 2048 + (size_t)tid * 16;
        asm volatile("global_load_async_to_lds_b128 %0, %1, off"
                     :: "v"(ldsOff), "v"(src) : "memory");
        asm volatile("global_load_async_to_lds_b128 %0, %1, off offset:16"
                     :: "v"(ldsOff), "v"(src) : "memory");
    };

    float sume[8] = {0, 0, 0, 0, 0, 0, 0, 0};
    float poss[8] = {0, 0, 0, 0, 0, 0, 0, 0};

    issueTile(0, 0);
    for (int nb = 0; nb < 800; ++nb) {
        asm volatile("s_wait_asynccnt 0x0" ::: "memory");  // our copies done
        __syncthreads();                                    // everyone's done
        if (nb + 1 < 800) issueTile(nb + 1, (nb + 1) & 1);  // overlap next DMA

        const __bf16* tile = &ldsT[nb & 1][0];
        int nlab = labAll[nb * 16 + (lane & 15)];
        v8f acc = {};
        #pragma unroll
        for (int t = 0; t < 4; ++t) {
            const __bf16* bp = tile + t * 512 + fragoff;
            FragBF fb;
            fb.u[0] = *(const uint4*)(bp + g * 8);         // ds_load_b128
            fb.u[1] = *(const uint4*)(bp + 16 + g * 8);   // ds_load_b128
            acc = __builtin_amdgcn_wmma_f32_16x16x32_bf16(false, av[t], false, fb.v,
                                                          (short)0, acc, false, false);
        }
        #pragma unroll
        for (int r = 0; r < 8; ++r) {
            float xv = acc[r] * 2.0f;   // 1/TEMP
            sume[r] += __expf(xv);      // |logit| bounded (<~23): no max shift needed
            if (nlab == rowlab[r]) poss[r] += xv;
        }
    }
    // reduce across the 16 lanes of each half (rows stay within half)
    #pragma unroll
    for (int msk = 8; msk >= 1; msk >>= 1)
        #pragma unroll
        for (int r = 0; r < 8; ++r) {
            sume[r] += __shfl_xor(sume[r], msk, 32);
            poss[r] += __shfl_xor(poss[r], msk, 32);
        }
    if ((lane & 15) == 0) {
        #pragma unroll
        for (int r = 0; r < 8; ++r) {
            int a = mbase + mofs + r;
            float cp = fmaxf((float)cntAll[rowlab[r]], 1.0f);
            float per = -(poss[r] - cp * __logf(sume[r])) / cp;
            atomicAdd(&patchSum[a / 100], per);
        }
    }
}

// ---------------------------------------------------------------------------
// final weighted sum (device-side epoch branch keeps launch deterministic)
// ---------------------------------------------------------------------------
__global__ void k_finalize(const float* __restrict__ accum,
                           const float* __restrict__ patchSum,
                           const float* __restrict__ patchLab,
                           const int* __restrict__ epoch,
                           float* __restrict__ out) {
    if (threadIdx.x != 0) return;
    float sup = accum[3] / fmaxf(accum[4], 1.f);
    int ep = epoch[0];
    if (ep < 5) { out[0] = sup; return; }
    float csum = 0.f, cinc = 0.f;
    for (int p = 0; p < 64; ++p) {
        float fc = patchLab[p] * 0.01f;
        float inc = ((fc > 0.1f) && (fc < 0.9f)) ? 0.f : 1.f;
        csum += inc * (patchSum[p] * 0.01f);
        cinc += inc;
    }
    float contr = csum / fmaxf(cinc, 1.f);
    float ent   = accum[0] / 6400.f;
    float cons  = accum[1] / fmaxf(accum[2], 1.f);
    float ramp  = fminf(fmaxf(ep / 40.f, 0.f), 1.f);
    float cw    = 1.0f * __expf(-5.f * (1.f - ramp) * (1.f - ramp));
    out[0] = sup + 0.1f * contr + cw * cons + 0.01f * ent;
}

// ---------------------------------------------------------------------------
// host orchestration
// ---------------------------------------------------------------------------
static inline size_t cdiv(size_t a, size_t b) { return (a + b - 1) / b; }

extern "C" void kernel_launch(void* const* d_in, const int* in_sizes, int n_in,
                              void* d_out, int out_size, void* d_ws, size_t ws_size,
                              hipStream_t stream) {
    (void)in_sizes; (void)n_in; (void)out_size; (void)ws_size;
    const float* x_l   = (const float*)d_in[0];
    const int*   y_l   = (const int*)d_in[1];
    const float* x_ul  = (const float*)d_in[2];
    const int*   epoch = (const int*)d_in[3];
    const float* pema  = (const float*)d_in[4];
    const float* zema  = (const float*)d_in[5];
    const float* W1    = (const float*)d_in[6];
    const float* W2    = (const float*)d_in[7];
    const float* W3    = (const float*)d_in[8];
    const float* Wc    = (const float*)d_in[9];
    const float* bc    = (const float*)d_in[10];
    const float* Wp    = (const float*)d_in[11];
    float* out = (float*)d_out;

    // workspace carve (static offsets, ~109 MB total).
    // bf16 GEMM operands get +1KB slack: the pipelined loop reads one
    // 32-half tile past the end.
    char* base = (char*)d_ws;
    size_t off = 0;
    auto carve = [&](size_t bytes) -> char* {
        off = (off + 255) & ~(size_t)255;
        char* p = base + off;
        off += bytes;
        return p;
    };
    const size_t SLACK = 1024;
    __bf16* W1t   = (__bf16*)carve((size_t)32 * 64 * 2 + SLACK);
    __bf16* W2t   = (__bf16*)carve((size_t)576 * 128 * 2 + SLACK);
    __bf16* W3t   = (__bf16*)carve((size_t)1152 * 256 * 2 + SLACK);
    __bf16* Wpt   = (__bf16*)carve((size_t)256 * 128 * 2 + SLACK);
    __bf16* A1    = (__bf16*)carve((size_t)102400 * 32 * 2 + SLACK);
    float*  H1    = (float*)carve((size_t)102400 * 64 * 4);
    __bf16* A2    = (__bf16*)carve((size_t)25600 * 576 * 2 + SLACK);
    float*  H2    = (float*)carve((size_t)25600 * 128 * 4);
    __bf16* A3    = (__bf16*)carve((size_t)6400 * 1152 * 2 + SLACK);
    float*  FEA   = (float*)carve((size_t)6400 * 256 * 4);
    __bf16* FEAb  = (__bf16*)carve((size_t)6400 * 256 * 2 + SLACK);
    float*  PROJ  = (float*)carve((size_t)6400 * 128 * 4);
    __bf16* Anc   = (__bf16*)carve((size_t)6400 * 128 * 2 + SLACK);
    __bf16* FeatP = (__bf16*)carve((size_t)128 * 12800 * 2 + SLACK);
    float*  Zs    = (float*)carve((size_t)6400 * 2 * 4);
    float*  Zu    = (float*)carve((size_t)6400 * 2 * 4);
    int*    labAll  = (int*)carve((size_t)12800 * 4);
    int*    cntAll  = (int*)carve((size_t)2 * 4);
    float*  patchSum = (float*)carve((size_t)64 * 4);
    float*  patchLab = (float*)carve((size_t)64 * 4);
    float*  accum    = (float*)carve((size_t)8 * 4);

    hipLaunchKernelGGL(k_zero, dim3(1), dim3(256), 0, stream,
                       accum, cntAll, patchSum, patchLab);

    // weight packing (fragment-major)
    hipLaunchKernelGGL(k_pack_wt, dim3(cdiv(32 * 64, 256)), dim3(256), 0, stream,
                       W1, W1t, 64, 32, 27);
    hipLaunchKernelGGL(k_pack_wt, dim3(cdiv(576 * 128, 256)), dim3(256), 0, stream,
                       W2, W2t, 128, 576, 576);
    hipLaunchKernelGGL(k_pack_wt, dim3(cdiv(1152 * 256, 256)), dim3(256), 0, stream,
                       W3, W3t, 256, 1152, 1152);
    hipLaunchKernelGGL(k_pack_wt, dim3(cdiv(256 * 128, 256)), dim3(256), 0, stream,
                       Wp, Wpt, 128, 256, 256);

    // extractor as implicit GEMM (runs for x_l then x_ul)
    auto extractor = [&](const float* x) {
        hipLaunchKernelGGL(k_im2col, dim3(cdiv((size_t)102400 * 32, 256)), dim3(256), 0, stream,
                           x, A1, 4, 3, 320, 320, 160, 160, 32, 27, 1);
        hipLaunchKernelGGL((k_gemm_wmma<64, 32, true>), dim3(102400 / 64, 64 / 16), dim3(128), 0, stream,
                           A1, W1t, H1);
        hipLaunchKernelGGL(k_im2col, dim3(cdiv((size_t)25600 * 576, 256)), dim3(256), 0, stream,
                           H1, A2, 4, 64, 160, 160, 80, 80, 576, 576, 0);
        hipLaunchKernelGGL((k_gemm_wmma<128, 576, true>), dim3(25600 / 64, 128 / 16), dim3(128), 0, stream,
                           A2, W2t, H2);
        hipLaunchKernelGGL(k_im2col, dim3(cdiv((size_t)6400 * 1152, 256)), dim3(256), 0, stream,
                           H2, A3, 4, 128, 80, 80, 40, 40, 1152, 1152, 0);
        hipLaunchKernelGGL((k_gemm_wmma<256, 1152, true>), dim3(6400 / 64, 256 / 16), dim3(128), 0, stream,
                           A3, W3t, FEA);
    };

    // ---- labeled branch: supervised CE ----
    extractor(x_l);
    hipLaunchKernelGGL(k_head2, dim3(cdiv(6400, 256)), dim3(256), 0, stream,
                       FEA, Wc, bc, Zs, 6400);
    hipLaunchKernelGGL(k_sup_loss, dim3(409600 / 256), dim3(256), 0, stream,
                       Zs, y_l, accum);

    // ---- unlabeled branch ----
    extractor(x_ul);
    hipLaunchKernelGGL(k_head2, dim3(cdiv(6400, 256)), dim3(256), 0, stream,
                       FEA, Wc, bc, Zu, 6400);
    hipLaunchKernelGGL(k_f2bf, dim3(cdiv((size_t)6400 * 256, 256)), dim3(256), 0, stream,
                       FEA, FEAb, (size_t)6400 * 256);
    hipLaunchKernelGGL((k_gemm_wmma<128, 256, false>), dim3(6400 / 64, 128 / 16), dim3(128), 0, stream,
                       FEAb, Wpt, PROJ);
    hipLaunchKernelGGL(k_norm_pack, dim3(cdiv(6400, 256)), dim3(256), 0, stream,
                       PROJ, Anc, FeatP);
    hipLaunchKernelGGL(k_pack_ema, dim3(cdiv(6400, 256)), dim3(256), 0, stream,
                       pema, FeatP);
    hipLaunchKernelGGL(k_z_stats, dim3(6400 / 256), dim3(256), 0, stream,
                       Zu, zema, labAll, cntAll, patchLab, accum);
    hipLaunchKernelGGL(k_contr_wmma, dim3(6400 / 16 / 4), dim3(128), 0, stream,
                       Anc, FeatP, labAll, cntAll, patchSum);

    hipLaunchKernelGGL(k_finalize, dim3(1), dim3(32), 0, stream,
                       accum, patchSum, patchLab, epoch, out);
}
